// DecoderBlock_73203422593486
// MI455X (gfx1250) — matrix operand order repaired and checked by reading, not compiled, not run on hardware
//
#include <hip/hip_runtime.h>
#include <math.h>

typedef __bf16 bf16;
typedef __attribute__((ext_vector_type(8)))  bf16  bf16x8;
typedef __attribute__((ext_vector_type(16))) bf16  v16bf;
typedef __attribute__((ext_vector_type(8)))  float v8f;

#define D_MODEL 1024
#define FF_DIM  4096
#define BATCH   2
#define SEQ     2048
#define HEADS   16
#define HDIM    64
#define MTOK    (BATCH*SEQ)   /* 4096 tokens */

// ---------------------------------------------------------------- helpers
static __device__ inline bf16 f2bf(float f) {
    unsigned u = __float_as_uint(f);
    u += 0x7FFFu + ((u >> 16) & 1u);           // round-to-nearest-even
    unsigned short h = (unsigned short)(u >> 16);
    return __builtin_bit_cast(bf16, h);
}

// A-matrix fragment (16x32 bf16): lanes 0-15 hold K {0..7,16..23},
// lanes 16-31 hold K {8..15,24..31}  (ISA 7.12.2 16-bit A layout)
static __device__ inline v16bf load_fragA(const bf16* rowp, int kbase, int hl) {
    bf16x8 lo = *(const bf16x8*)(rowp + kbase + 8*hl);
    bf16x8 hi = *(const bf16x8*)(rowp + kbase + 8*hl + 16);
    v16bf r;
#pragma unroll
    for (int i = 0; i < 8; ++i) { r[i] = lo[i]; r[i+8] = hi[i]; }
    return r;
}

// B-matrix fragment (32x16 bf16): lanes 0-15 hold K {0..7,16..23}, lanes 16-31 K {8..15,24..31}
static __device__ inline v16bf load_fragB(const bf16* rowp, int kbase, int hl) {
    bf16x8 lo = *(const bf16x8*)(rowp + kbase + 8*hl);
    bf16x8 hi = *(const bf16x8*)(rowp + kbase + 8*hl + 16);
    v16bf r;
#pragma unroll
    for (int i = 0; i < 8; ++i) { r[i] = lo[i]; r[i+8] = hi[i]; }
    return r;
}

static __device__ inline v8f wmma_bf16(v16bf a, v16bf b, v8f c) {
    return __builtin_amdgcn_wmma_f32_16x16x32_bf16(false, a, false, b,
                                                   (short)0, c, false, false);
}

// CDNA5 async global->LDS copy (ASYNCcnt path, GVS addressing mode):
//   lds[lane.lds_off] = mem[saddr + lane.gbyte_off], 16 bytes per lane
static __device__ inline void async_b128(unsigned lds_off, unsigned gbyte_off,
                                         const void* sbase) {
    asm volatile("global_load_async_to_lds_b128 %0, %1, %2"
                 :: "v"(lds_off), "v"(gbyte_off), "s"(sbase) : "memory");
}
static __device__ inline void wait_async0() {
    asm volatile("s_wait_asynccnt 0x0" ::: "memory");
}

#define V8F_ZERO {0.f,0.f,0.f,0.f,0.f,0.f,0.f,0.f}

// ---------------------------------------------------------------- fp32 -> bf16
__global__ void cvt_kernel(const float* __restrict__ in, bf16* __restrict__ out, size_t n) {
    size_t i = (size_t)blockIdx.x * blockDim.x + threadIdx.x;
    size_t stride = (size_t)gridDim.x * blockDim.x;
    for (; i < n; i += stride) out[i] = f2bf(in[i]);
}

// ---------------------------------------------------------------- GEMM: C = A*W^T + bias
// A: [M,K] bf16 row-major, W: [N,K] bf16 row-major (torch Linear weight layout)
#define GBM 128
#define GBN 128
#define GBK 32
#define GLS 40   /* padded LDS row stride (halfwords) -> conflict-free b128 frag loads */

template<bool GELU, bool OUTF, bool OUTB>
__global__ __launch_bounds__(256) void gemm_kernel(
    const bf16* __restrict__ A, const bf16* __restrict__ W,
    const float* __restrict__ bias, float* __restrict__ Cf, bf16* __restrict__ Cb,
    int M, int N, int K)
{
    __shared__ bf16 As[2][GBM * GLS];
    __shared__ bf16 Ws[2][GBN * GLS];
    const int t    = threadIdx.x;
    const int lane = t & 31, wid = t >> 5;
    const int hl   = lane >> 4, l15 = lane & 15;
    const int wm   = wid >> 2, wn = wid & 3;            // 2x4 wave grid
    const int bM   = blockIdx.y * GBM, bN = blockIdx.x * GBN;

    v8f acc[4][2];
#pragma unroll
    for (int i = 0; i < 4; ++i)
#pragma unroll
        for (int j = 0; j < 2; ++j) acc[i][j] = (v8f)V8F_ZERO;

    // staging assignment: thread t handles tile row (t>>1), 16-elem chunk pair at col (t&1)*16
    const int ldr = t >> 1;
    const int ldc = (t & 1) * 16;
    const unsigned goffA = (unsigned)((((size_t)(bM + ldr)) * K + ldc) * 2u);
    const unsigned goffW = (unsigned)((((size_t)(bN + ldr)) * K + ldc) * 2u);
    const unsigned ldsA0 = (unsigned)(uintptr_t)&As[0][ldr * GLS + ldc];
    const unsigned ldsA1 = (unsigned)(uintptr_t)&As[1][ldr * GLS + ldc];
    const unsigned ldsW0 = (unsigned)(uintptr_t)&Ws[0][ldr * GLS + ldc];
    const unsigned ldsW1 = (unsigned)(uintptr_t)&Ws[1][ldr * GLS + ldc];

    // prologue: stage k-tile 0 into buffer 0 (async, 16B per lane per op)
    async_b128(ldsA0,      goffA,      A);
    async_b128(ldsA0 + 16, goffA + 16, A);
    async_b128(ldsW0,      goffW,      W);
    async_b128(ldsW0 + 16, goffW + 16, W);

    const int nK = K / GBK;
    int p = 0;
    for (int i = 0; i < nK; ++i) {
        wait_async0();        // own async stores for tile i complete
        __syncthreads();      // tile i visible; everyone done reading buffer p^1
        if (i + 1 < nK) {     // prefetch tile i+1 into the other buffer (overlaps WMMAs)
            const unsigned gb = (unsigned)((i + 1) * (GBK * 2));
            const unsigned la = p ? ldsA0 : ldsA1;
            const unsigned lw = p ? ldsW0 : ldsW1;
            async_b128(la,      goffA + gb,      A);
            async_b128(la + 16, goffA + gb + 16, A);
            async_b128(lw,      goffW + gb,      W);
            async_b128(lw + 16, goffW + gb + 16, W);
        }
        const bf16* curA = As[p];
        const bf16* curW = Ws[p];
        v16bf aF[4], bF[2];
#pragma unroll
        for (int mt = 0; mt < 4; ++mt)
            aF[mt] = load_fragA(curA + (wm*64 + mt*16 + l15) * GLS, 0, hl);
#pragma unroll
        for (int nt = 0; nt < 2; ++nt)
            bF[nt] = load_fragB(curW + (wn*32 + nt*16 + l15) * GLS, 0, hl);
#pragma unroll
        for (int mt = 0; mt < 4; ++mt)
#pragma unroll
            for (int nt = 0; nt < 2; ++nt)
                acc[mt][nt] = wmma_bf16(aF[mt], bF[nt], acc[mt][nt]);
        p ^= 1;
    }

#pragma unroll
    for (int nt = 0; nt < 2; ++nt) {
        const int nG = bN + wn*32 + nt*16 + l15;
        const float bv = bias[nG];
#pragma unroll
        for (int mt = 0; mt < 4; ++mt) {
            const int mBase = bM + wm*64 + mt*16 + 8*hl;
#pragma unroll
            for (int r = 0; r < 8; ++r) {
                float v = acc[mt][nt][r] + bv;
                if (GELU) v = 0.5f * v * (1.f + erff(v * 0.70710678118654752440f));
                const size_t idx = (size_t)(mBase + r) * (size_t)N + (size_t)nG;
                if (OUTF) Cf[idx] = v;
                if (OUTB) Cb[idx] = f2bf(v);
            }
        }
    }
}

// ---------------------------------------------------------------- flash attention
// Q/K/V/O in [b, s, h*64+e] bf16 layout, per-(b,h) attention with online softmax
#define ATM 128
#define ATN 128
#define KSS 72
#define VTS 136
#define PSS 136

template<bool CAUSAL>
__global__ __launch_bounds__(256) void attn_kernel(
    const bf16* __restrict__ Qm, const bf16* __restrict__ Km,
    const bf16* __restrict__ Vm, bf16* __restrict__ Om)
{
    __shared__ bf16 Ks[ATN  * KSS];   // [key][hd]
    __shared__ bf16 Vt[HDIM * VTS];   // [hd][key] (transposed)
    __shared__ bf16 Ps[ATM  * PSS];   // probs, per-wave 16-row slabs

    const int t    = threadIdx.x;
    const int lane = t & 31, wid = t >> 5;
    const int hl   = lane >> 4, l15 = lane & 15;
    const int b    = blockIdx.y >> 4, h = blockIdx.y & 15;
    const int q0   = blockIdx.x * ATM;

    const size_t headOff = (size_t)b * SEQ * D_MODEL + (size_t)h * HDIM;

    // this wave's 16 query rows -> two A-fragments over HD=64
    const bf16* qrow = Qm + headOff + (size_t)(q0 + wid*16 + l15) * D_MODEL;
    v16bf aQ[2];
#pragma unroll
    for (int ks = 0; ks < 2; ++ks) aQ[ks] = load_fragA(qrow, ks*32, hl);

    float rowM[8], rowL[8];
    v8f oacc[4];
#pragma unroll
    for (int r = 0; r < 8; ++r) { rowM[r] = -__builtin_inff(); rowL[r] = 0.f; }
#pragma unroll
    for (int i = 0; i < 4; ++i) oacc[i] = (v8f)V8F_ZERO;

    const int nIter = CAUSAL ? (blockIdx.x + 1) : (SEQ / ATN);
    for (int it = 0; it < nIter; ++it) {
        const int k0 = it * ATN;
        __syncthreads();
        {   // stage K tile [128 x 64]
            const int row = t >> 1, c0 = (t & 1) * 32;
            const bf16* src = Km + headOff + (size_t)(k0 + row) * D_MODEL + c0;
            bf16* dst = Ks + row * KSS + c0;
#pragma unroll
            for (int i = 0; i < 4; ++i)
                *(bf16x8*)(dst + i*8) = *(const bf16x8*)(src + i*8);
        }
        {   // stage V tile transposed: Vt[hd][key]
#pragma unroll
            for (int cc = 0; cc < 4; ++cc) {
                const int c = t*4 + cc;
                const int key = c >> 3, hd0 = (c & 7) * 8;
                const bf16x8 v = *(const bf16x8*)(Vm + headOff +
                                   (size_t)(k0 + key) * D_MODEL + hd0);
#pragma unroll
                for (int j = 0; j < 8; ++j) Vt[(hd0 + j) * VTS + key] = v[j];
            }
        }
        __syncthreads();

        // scores S = Q K^T  (16 x 128 per wave)
        v8f sacc[8];
#pragma unroll
        for (int nt = 0; nt < 8; ++nt) sacc[nt] = (v8f)V8F_ZERO;
#pragma unroll
        for (int nt = 0; nt < 8; ++nt)
#pragma unroll
            for (int ks = 0; ks < 2; ++ks) {
                v16bf bK = load_fragB(Ks + (nt*16 + l15) * KSS, ks*32, hl);
                sacc[nt] = wmma_bf16(aQ[ks], bK, sacc[nt]);
            }

#pragma unroll
        for (int nt = 0; nt < 8; ++nt)
#pragma unroll
            for (int r = 0; r < 8; ++r) sacc[nt][r] *= 0.03125f;  // 1/sqrt(d_model)

        if (CAUSAL && it == blockIdx.x) {
#pragma unroll
            for (int nt = 0; nt < 8; ++nt) {
                const int nG = k0 + nt*16 + l15;
#pragma unroll
                for (int r = 0; r < 8; ++r) {
                    const int mG = q0 + wid*16 + r + 8*hl;
                    if (nG > mG) sacc[nt][r] = -__builtin_inff();
                }
            }
        }

        // ---- online softmax (rows live in 16-lane halves; xor<16 stays in-half)
        float tmax[8];
#pragma unroll
        for (int r = 0; r < 8; ++r) {
            tmax[r] = -__builtin_inff();
#pragma unroll
            for (int nt = 0; nt < 8; ++nt) tmax[r] = fmaxf(tmax[r], sacc[nt][r]);
        }
#pragma unroll
        for (int r = 0; r < 8; ++r)
#pragma unroll
            for (int m = 1; m < 16; m <<= 1)
                tmax[r] = fmaxf(tmax[r], __shfl_xor(tmax[r], m, 32));

        float scl[8], rsum[8];
#pragma unroll
        for (int r = 0; r < 8; ++r) {
            const float nM = fmaxf(rowM[r], tmax[r]);
            scl[r] = __expf(rowM[r] - nM);
            rowM[r] = nM;
            rsum[r] = 0.f;
        }
#pragma unroll
        for (int nt = 0; nt < 8; ++nt)
#pragma unroll
            for (int r = 0; r < 8; ++r) {
                const float p = __expf(sacc[nt][r] - rowM[r]);
                sacc[nt][r] = p;
                rsum[r] += p;
            }
#pragma unroll
        for (int r = 0; r < 8; ++r)
#pragma unroll
            for (int m = 1; m < 16; m <<= 1)
                rsum[r] += __shfl_xor(rsum[r], m, 32);
#pragma unroll
        for (int r = 0; r < 8; ++r) rowL[r] = rowL[r] * scl[r] + rsum[r];
#pragma unroll
        for (int o = 0; o < 4; ++o)
#pragma unroll
            for (int r = 0; r < 8; ++r) oacc[o][r] *= scl[r];

        // probs -> LDS (per-wave slab), re-shaped into A-frag layout for P@V
#pragma unroll
        for (int nt = 0; nt < 8; ++nt)
#pragma unroll
            for (int r = 0; r < 8; ++r)
                Ps[(wid*16 + r + 8*hl) * PSS + nt*16 + l15] = f2bf(sacc[nt][r]);
        asm volatile("s_wait_dscnt 0x0" ::: "memory");   // same-wave cross-lane LDS RAW

        // O += P @ V   (K dim = 128 keys, N dim = 64 hd)
#pragma unroll
        for (int ks2 = 0; ks2 < 4; ++ks2) {
            v16bf aP = load_fragA(Ps + (wid*16 + l15) * PSS, ks2*32, hl);
#pragma unroll
            for (int nt2 = 0; nt2 < 4; ++nt2) {
                v16bf bV = load_fragB(Vt + (nt2*16 + l15) * VTS, ks2*32, hl);
                oacc[nt2] = wmma_bf16(aP, bV, oacc[nt2]);
            }
        }
    }

    float inv[8];
#pragma unroll
    for (int r = 0; r < 8; ++r) inv[r] = 1.f / rowL[r];
#pragma unroll
    for (int nt2 = 0; nt2 < 4; ++nt2)
#pragma unroll
        for (int r = 0; r < 8; ++r) {
            const size_t idx = headOff +
                (size_t)(q0 + wid*16 + r + 8*hl) * D_MODEL + nt2*16 + l15;
            Om[idx] = f2bf(oacc[nt2][r] * inv[r]);
        }
}

// ---------------------------------------------------------------- residual + LayerNorm
__global__ __launch_bounds__(256) void ln_kernel(
    const float* __restrict__ X, const float* __restrict__ R,
    const float* __restrict__ G, const float* __restrict__ Bb,
    float* __restrict__ Yf, bf16* __restrict__ Yb)
{
    __shared__ float red[16];
    const int row = blockIdx.x, t = threadIdx.x;
    const int lane = t & 31, wid = t >> 5;
    const size_t base = (size_t)row * D_MODEL;
    float v[4], s = 0.f, s2 = 0.f;
#pragma unroll
    for (int i = 0; i < 4; ++i) {
        const int c = t + i * 256;
        const float val = X[base + c] + R[base + c];
        v[i] = val; s += val; s2 += val * val;
    }
#pragma unroll
    for (int m = 1; m < 32; m <<= 1) {
        s  += __shfl_xor(s,  m, 32);
        s2 += __shfl_xor(s2, m, 32);
    }
    if (lane == 0) { red[wid] = s; red[8 + wid] = s2; }
    __syncthreads();
    float S = 0.f, S2 = 0.f;
#pragma unroll
    for (int w = 0; w < 8; ++w) { S += red[w]; S2 += red[8 + w]; }
    const float mean = S * (1.f / D_MODEL);
    const float var  = S2 * (1.f / D_MODEL) - mean * mean;
    const float rstd = rsqrtf(var + 1e-5f);
#pragma unroll
    for (int i = 0; i < 4; ++i) {
        const int c = t + i * 256;
        const float y = (v[i] - mean) * rstd * G[c] + Bb[c];
        Yf[base + c] = y;
        if (Yb) Yb[base + c] = f2bf(y);
    }
}

// ---------------------------------------------------------------- host orchestration
extern "C" void kernel_launch(void* const* d_in, const int* in_sizes, int n_in,
                              void* d_out, int out_size, void* d_ws, size_t ws_size,
                              hipStream_t stream)
{
    (void)in_sizes; (void)n_in; (void)out_size; (void)ws_size;

    const float* x      = (const float*)d_in[0];
    const float* enc    = (const float*)d_in[1];
    const float* sa_q_W = (const float*)d_in[2];  const float* sa_q_b = (const float*)d_in[3];
    const float* sa_k_W = (const float*)d_in[4];  const float* sa_k_b = (const float*)d_in[5];
    const float* sa_v_W = (const float*)d_in[6];  const float* sa_v_b = (const float*)d_in[7];
    const float* ca_q_W = (const float*)d_in[8];  const float* ca_q_b = (const float*)d_in[9];
    const float* ca_k_W = (const float*)d_in[10]; const float* ca_k_b = (const float*)d_in[11];
    const float* ca_v_W = (const float*)d_in[12]; const float* ca_v_b = (const float*)d_in[13];
    const float* m_Wq   = (const float*)d_in[14]; const float* m_bq   = (const float*)d_in[15];
    const float* m_Wk   = (const float*)d_in[16]; const float* m_bk   = (const float*)d_in[17];
    const float* m_Wv   = (const float*)d_in[18]; const float* m_bv   = (const float*)d_in[19];
    const float* m_Wo   = (const float*)d_in[20]; const float* m_bo   = (const float*)d_in[21];
    const float* c_Wq   = (const float*)d_in[22]; const float* c_bq   = (const float*)d_in[23];
    const float* c_Wk   = (const float*)d_in[24]; const float* c_bk   = (const float*)d_in[25];
    const float* c_Wv   = (const float*)d_in[26]; const float* c_bv   = (const float*)d_in[27];
    const float* c_Wo   = (const float*)d_in[28]; const float* c_bo   = (const float*)d_in[29];
    const float* ff_W1  = (const float*)d_in[30]; const float* ff_b1  = (const float*)d_in[31];
    const float* ff_W2  = (const float*)d_in[32]; const float* ff_b2  = (const float*)d_in[33];
    const float* ln1_g  = (const float*)d_in[34]; const float* ln1_b  = (const float*)d_in[35];
    const float* ln2_g  = (const float*)d_in[36]; const float* ln2_b  = (const float*)d_in[37];
    const float* ln3_g  = (const float*)d_in[38]; const float* ln3_b  = (const float*)d_in[39];

    char* ws = (char*)d_ws;
    size_t off = 0;
    auto alloc = [&](size_t bytes) -> char* {
        off = (off + 255) & ~(size_t)255;
        char* p = ws + off;
        off += bytes;
        return p;
    };

    const size_t DD  = (size_t)D_MODEL * D_MODEL;     // 1M
    const size_t DF  = (size_t)D_MODEL * FF_DIM;      // 4M
    const size_t TD  = (size_t)MTOK * D_MODEL;        // 4M
    const size_t TF  = (size_t)MTOK * FF_DIM;         // 16M

    // bf16 weight copies
    bf16* w_saq = (bf16*)alloc(DD*2); bf16* w_sak = (bf16*)alloc(DD*2); bf16* w_sav = (bf16*)alloc(DD*2);
    bf16* w_mq  = (bf16*)alloc(DD*2); bf16* w_mk  = (bf16*)alloc(DD*2); bf16* w_mv  = (bf16*)alloc(DD*2);
    bf16* w_mo  = (bf16*)alloc(DD*2);
    bf16* w_caq = (bf16*)alloc(DD*2); bf16* w_cak = (bf16*)alloc(DD*2); bf16* w_cav = (bf16*)alloc(DD*2);
    bf16* w_cq  = (bf16*)alloc(DD*2); bf16* w_ck  = (bf16*)alloc(DD*2); bf16* w_cv  = (bf16*)alloc(DD*2);
    bf16* w_co  = (bf16*)alloc(DD*2);
    bf16* w_f1  = (bf16*)alloc(DF*2); bf16* w_f2  = (bf16*)alloc(DF*2);
    // activations
    bf16* xb  = (bf16*)alloc(TD*2);  bf16* eb  = (bf16*)alloc(TD*2);
    bf16* q1  = (bf16*)alloc(TD*2);  bf16* k1  = (bf16*)alloc(TD*2);  bf16* v1 = (bf16*)alloc(TD*2);
    bf16* q2  = (bf16*)alloc(TD*2);  bf16* k2  = (bf16*)alloc(TD*2);  bf16* v2 = (bf16*)alloc(TD*2);
    bf16* ao  = (bf16*)alloc(TD*2);
    float* brf = (float*)alloc(TD*4);
    float* x1f = (float*)alloc(TD*4); bf16* x1b = (bf16*)alloc(TD*2);
    float* x2f = (float*)alloc(TD*4); bf16* x2b = (bf16*)alloc(TD*2);
    bf16* hb  = (bf16*)alloc(TF*2);
    float* out = (float*)d_out;

    auto cvt = [&](const float* in, bf16* o, size_t n) {
        cvt_kernel<<<dim3(2048), dim3(256), 0, stream>>>(in, o, n);
    };
    // bf16-output GEMM (optionally exact-erf GELU), fp32-output GEMM
    auto gemm_b = [&](const bf16* A, const bf16* W, const float* bias,
                      bf16* Cb, int M, int N, int K, int gelu) {
        dim3 g(N/GBN, M/GBM), blk(256);
        if (gelu) gemm_kernel<true,  false, true><<<g, blk, 0, stream>>>(A, W, bias, nullptr, Cb, M, N, K);
        else      gemm_kernel<false, false, true><<<g, blk, 0, stream>>>(A, W, bias, nullptr, Cb, M, N, K);
    };
    auto gemm_f = [&](const bf16* A, const bf16* W, const float* bias,
                      float* Cf, int M, int N, int K) {
        dim3 g(N/GBN, M/GBM), blk(256);
        gemm_kernel<false, true, false><<<g, blk, 0, stream>>>(A, W, bias, Cf, nullptr, M, N, K);
    };
    auto attn = [&](const bf16* Q, const bf16* K_, const bf16* V_, bf16* O_, int causal) {
        dim3 g(SEQ/ATM, BATCH*HEADS), blk(256);
        if (causal) attn_kernel<true ><<<g, blk, 0, stream>>>(Q, K_, V_, O_);
        else        attn_kernel<false><<<g, blk, 0, stream>>>(Q, K_, V_, O_);
    };
    auto ln = [&](const float* X, const float* R, const float* g_, const float* b_,
                  float* Yf, bf16* Yb) {
        ln_kernel<<<dim3(MTOK), dim3(256), 0, stream>>>(X, R, g_, b_, Yf, Yb);
    };

    // ---- weight / input conversion
    cvt(sa_q_W, w_saq, DD); cvt(sa_k_W, w_sak, DD); cvt(sa_v_W, w_sav, DD);
    cvt(m_Wq,  w_mq,  DD);  cvt(m_Wk,  w_mk,  DD);  cvt(m_Wv,  w_mv,  DD);
    cvt(m_Wo,  w_mo,  DD);
    cvt(ca_q_W, w_caq, DD); cvt(ca_k_W, w_cak, DD); cvt(ca_v_W, w_cav, DD);
    cvt(c_Wq,  w_cq,  DD);  cvt(c_Wk,  w_ck,  DD);  cvt(c_Wv,  w_cv,  DD);
    cvt(c_Wo,  w_co,  DD);
    cvt(ff_W1, w_f1,  DF);  cvt(ff_W2, w_f2,  DF);
    cvt(x,   xb, TD);       cvt(enc, eb, TD);

    // ---- self attention
    gemm_b(xb, w_saq, sa_q_b, q1, MTOK, D_MODEL, D_MODEL, 0);
    gemm_b(xb, w_sak, sa_k_b, k1, MTOK, D_MODEL, D_MODEL, 0);
    gemm_b(xb, w_sav, sa_v_b, v1, MTOK, D_MODEL, D_MODEL, 0);
    gemm_b(q1, w_mq,  m_bq,  q2, MTOK, D_MODEL, D_MODEL, 0);
    gemm_b(k1, w_mk,  m_bk,  k2, MTOK, D_MODEL, D_MODEL, 0);
    gemm_b(v1, w_mv,  m_bv,  v2, MTOK, D_MODEL, D_MODEL, 0);
    attn(q2, k2, v2, ao, 1);
    gemm_f(ao, w_mo,  m_bo,  brf, MTOK, D_MODEL, D_MODEL);
    ln(x, brf, ln1_g, ln1_b, x1f, x1b);

    // ---- cross attention
    gemm_b(x1b, w_caq, ca_q_b, q1, MTOK, D_MODEL, D_MODEL, 0);
    gemm_b(eb,  w_cak, ca_k_b, k1, MTOK, D_MODEL, D_MODEL, 0);
    gemm_b(eb,  w_cav, ca_v_b, v1, MTOK, D_MODEL, D_MODEL, 0);
    gemm_b(q1,  w_cq,  c_bq,  q2, MTOK, D_MODEL, D_MODEL, 0);
    gemm_b(k1,  w_ck,  c_bk,  k2, MTOK, D_MODEL, D_MODEL, 0);
    gemm_b(v1,  w_cv,  c_bv,  v2, MTOK, D_MODEL, D_MODEL, 0);
    attn(q2, k2, v2, ao, 0);
    gemm_f(ao,  w_co,  c_bo,  brf, MTOK, D_MODEL, D_MODEL);
    ln(x1f, brf, ln2_g, ln2_b, x2f, x2b);

    // ---- feed forward
    gemm_b(x2b, w_f1, ff_b1, hb, MTOK, FF_DIM, D_MODEL, 1);
    gemm_f(hb,  w_f2, ff_b2, brf, MTOK, D_MODEL, FF_DIM);
    ln(x2f, brf, ln3_g, ln3_b, out, nullptr);
}
